// RWKV_TimeMix_39565238731236
// MI455X (gfx1250) — matrix operand order repaired and checked by reading, not compiled, not run on hardware
//
#include <hip/hip_runtime.h>
#include <cstdint>
#include <cstddef>

#define Bq 8
#define Tq 1024
#define Cq 1024
#define CH 16          // scan chunks per sequence
#define LCH (Tq / CH)  // 64 timesteps per chunk

typedef __attribute__((ext_vector_type(16))) _Float16 v16h;
typedef __attribute__((ext_vector_type(8)))  _Float16 v8h;
typedef __attribute__((ext_vector_type(8)))  float    v8f;
typedef __attribute__((ext_vector_type(4)))  int      v4i;

// ---------------------------------------------------------------------------
// Async global->LDS copy (CDNA5 ASYNCcnt path) with portable fallback.
// ---------------------------------------------------------------------------
#if defined(__AMDGCN__) && __has_builtin(__builtin_amdgcn_global_load_async_to_lds_b128)
#define HAVE_ASYNC 1
#endif

__device__ __forceinline__ void async_copy16(const _Float16* g, _Float16* l) {
#ifdef HAVE_ASYNC
  __builtin_amdgcn_global_load_async_to_lds_b128(
      (__attribute__((address_space(1))) v4i*)g,
      (__attribute__((address_space(3))) v4i*)l, 0, 0);
#else
  *(v8h*)l = *(const v8h*)g;
#endif
}

#ifdef HAVE_ASYNC
#if __has_builtin(__builtin_amdgcn_s_wait_asynccnt)
#define WAIT_ASYNC(n) __builtin_amdgcn_s_wait_asynccnt(n)
#else
#define WAIT_ASYNC(n) asm volatile("s_wait_asynccnt %0" ::"i"(n) : "memory")
#endif
#else
#define WAIT_ASYNC(n) ((void)0)
#endif

// ---------------------------------------------------------------------------
// Convert the four weight matrices f32 -> f16 in one pass.
// ---------------------------------------------------------------------------
__global__ __launch_bounds__(256) void cvt_w4(const float* __restrict__ w0, const float* __restrict__ w1,
                                              const float* __restrict__ w2, const float* __restrict__ w3,
                                              _Float16* __restrict__ h0, _Float16* __restrict__ h1,
                                              _Float16* __restrict__ h2, _Float16* __restrict__ h3, int n) {
  int i = blockIdx.x * 256 + threadIdx.x;
  if (i < n) {
    h0[i] = (_Float16)w0[i];
    h1[i] = (_Float16)w1[i];
    h2[i] = (_Float16)w2[i];
    h3[i] = (_Float16)w3[i];
  }
}

// ---------------------------------------------------------------------------
// Time-shift + mix, output f16 activations for the WMMA GEMMs.
// ---------------------------------------------------------------------------
__global__ __launch_bounds__(256) void mix_kernel(const float* __restrict__ x,
                                                  const float* __restrict__ mk,
                                                  const float* __restrict__ mv,
                                                  const float* __restrict__ mr,
                                                  _Float16* __restrict__ xk,
                                                  _Float16* __restrict__ xv,
                                                  _Float16* __restrict__ xr) {
  int i = blockIdx.x * 256 + threadIdx.x;  // over B*T*C
  int c = i & (Cq - 1);
  int t = (i >> 10) & (Tq - 1);
  float xi = x[i];
  float xp = (t != 0) ? x[i - Cq] : 0.0f;
  float a = mk[c];
  float b = mv[c];
  float r = mr[c];
  xk[i] = (_Float16)(xi * a + xp * (1.0f - a));
  xv[i] = (_Float16)(xi * b + xp * (1.0f - b));
  xr[i] = (_Float16)(xi * r + xp * (1.0f - r));
}

// ---------------------------------------------------------------------------
// C[M,N](f32) = A[M,K](f16,row-major) * W[N,K]^T (f16,row-major)
// Block 128 thr = 4 waves (2x2); wave tile 64x64 = 4x4 WMMA; block tile
// 128x128; K-step 32; LDS double-buffered, staged via async-to-LDS.
// Steady-state: 16 ds_load_b128 : 16 v_wmma per wave per K-step.
// ---------------------------------------------------------------------------
#define BLK_K 32
#define PITCH 40  // halves per LDS row (32 data + 8 pad), 16B-aligned rows

__global__ __launch_bounds__(128) void gemm_wmma_lds(const _Float16* __restrict__ A,
                                                     const _Float16* __restrict__ W,
                                                     float* __restrict__ Cout,
                                                     int M, int N, int K) {
  __shared__ _Float16 As[2][128 * PITCH];
  __shared__ _Float16 Bs[2][128 * PITCH];

  const int tid = threadIdx.x;
  const int lane = tid & 31;
  const int wave = tid >> 5;
  const int wm = wave >> 1;  // 0..1
  const int wn = wave & 1;   // 0..1
  const int lr = lane & 15;
  const int lh = lane >> 4;
  const int row0b = blockIdx.y * 128;
  const int col0b = blockIdx.x * 128;

  v8f acc[4][4];
#pragma unroll
  for (int mi = 0; mi < 4; ++mi)
#pragma unroll
    for (int ni = 0; ni < 4; ++ni)
#pragma unroll
      for (int j = 0; j < 8; ++j) acc[mi][ni][j] = 0.0f;

  // Stage one 128x32 A tile + one 128x32 B tile: 512 16B-chunks each,
  // 4 A-chunks + 4 B-chunks per thread (4 chunks per row of 32 halves).
  auto stage = [&](int buf, int k0) {
#pragma unroll
    for (int i = 0; i < 4; ++i) {
      int id = tid + i * 128;
      int r = id >> 2, cc = (id & 3) * 8;  // cc in halves
      async_copy16(A + (size_t)(row0b + r) * K + k0 + cc, &As[buf][r * PITCH + cc]);
    }
#pragma unroll
    for (int i = 0; i < 4; ++i) {
      int id = tid + i * 128;
      int r = id >> 2, cc = (id & 3) * 8;
      async_copy16(W + (size_t)(col0b + r) * K + k0 + cc, &Bs[buf][r * PITCH + cc]);
    }
  };

  auto compute = [&](int buf) {
    const _Float16* as = As[buf];
    const _Float16* bs = Bs[buf];
    v16h bf[4];
#pragma unroll
    for (int ni = 0; ni < 4; ++ni) {
      // B layout: lane holds col N=lr, K chunk lh*16 .. +15 (contiguous)
      const _Float16* p = bs + (wn * 64 + ni * 16 + lr) * PITCH + lh * 16;
      ((v8h*)&bf[ni])[0] = *(const v8h*)(p);
      ((v8h*)&bf[ni])[1] = *(const v8h*)(p + 8);
    }
#pragma unroll
    for (int mi = 0; mi < 4; ++mi) {
      // A layout: lane holds row M=lr, K chunk lh*16 .. +15 (contiguous)
      v16h af;
      const _Float16* p = as + (wm * 64 + mi * 16 + lr) * PITCH + lh * 16;
      ((v8h*)&af)[0] = *(const v8h*)(p);
      ((v8h*)&af)[1] = *(const v8h*)(p + 8);
#pragma unroll
      for (int ni = 0; ni < 4; ++ni)
        acc[mi][ni] = __builtin_amdgcn_wmma_f32_16x16x32_f16(
            false, af, false, bf[ni], (short)0, acc[mi][ni], false, false);
    }
  };

  const int NK = K / BLK_K;
  stage(0, 0);
  for (int ks = 0; ks < NK - 1; ++ks) {
    stage((ks + 1) & 1, (ks + 1) * BLK_K);  // overlap next-tile copy with compute
    WAIT_ASYNC(8);                          // oldest 8 (= cur tile's copies) done
    __syncthreads();                        // cur tile visible to all waves
    compute(ks & 1);
    __syncthreads();                        // all done reading cur before overwrite
  }
  WAIT_ASYNC(0);
  __syncthreads();
  compute((NK - 1) & 1);

#pragma unroll
  for (int mi = 0; mi < 4; ++mi) {
#pragma unroll
    for (int ni = 0; ni < 4; ++ni) {
      float* o = Cout + (size_t)(row0b + wm * 64 + mi * 16 + lh * 8) * N + col0b + wn * 64 + ni * 16 + lr;
#pragma unroll
      for (int i = 0; i < 8; ++i) o[(size_t)i * N] = acc[mi][ni][i];
    }
  }
}

// ---------------------------------------------------------------------------
// WKV recurrence as a 3-pass chunked scan (16x more parallelism than 1-pass):
//   pass1: per-chunk local states (zero incoming)
//   pass2: combine chunk states with dw^LCH, store incoming state per chunk
//   pass3: re-scan each chunk from its incoming state, emit f16 rwkv
// ---------------------------------------------------------------------------
__global__ __launch_bounds__(256) void scan_part1(const float* __restrict__ kraw,
                                                  const float* __restrict__ vraw,
                                                  const float* __restrict__ td,
                                                  float* __restrict__ Aend, float* __restrict__ Bend) {
  int g = blockIdx.x * 256 + threadIdx.x;  // over CH * B*C (chunk-major)
  int bc = g & (Bq * Cq - 1);
  int j = g >> 13;  // / (B*C)
  int c = bc & (Cq - 1);
  int b = bc >> 10;
  float dw = __expf(-__expf(td[c]));
  float Ac = 0.0f, Bc = 0.0f;
  size_t base = (size_t)b * Tq * Cq + (size_t)j * LCH * Cq + c;
  for (int t = 0; t < LCH; ++t) {
    size_t off = base + (size_t)t * Cq;
    float ke = __expf(fminf(kraw[off], 60.0f));
    float kv = ke * vraw[off];
    Ac = kv + dw * Ac;
    Bc = ke + dw * Bc;
  }
  Aend[(size_t)bc * CH + j] = Ac;
  Bend[(size_t)bc * CH + j] = Bc;
}

__global__ __launch_bounds__(256) void scan_part2(const float* __restrict__ td,
                                                  float* __restrict__ Aend, float* __restrict__ Bend) {
  int bc = blockIdx.x * 256 + threadIdx.x;  // over B*C
  int c = bc & (Cq - 1);
  float dw = __expf(-__expf(td[c]));
  float dwp = dw;
#pragma unroll
  for (int i = 0; i < 6; ++i) dwp *= dwp;  // dw^64 == dw^LCH
  float Ain = 0.0f, Bin = 0.0f;
  for (int j = 0; j < CH; ++j) {
    size_t o = (size_t)bc * CH + j;
    float Ae = Aend[o], Be = Bend[o];
    Aend[o] = Ain;  // overwrite with incoming state for chunk j
    Bend[o] = Bin;
    Ain = Ae + dwp * Ain;
    Bin = Be + dwp * Bin;
  }
}

__global__ __launch_bounds__(256) void scan_part3(const float* __restrict__ kraw,
                                                  const float* __restrict__ vraw,
                                                  const float* __restrict__ rraw,
                                                  const float* __restrict__ td,
                                                  const float* __restrict__ tf,
                                                  const float* __restrict__ Ain,
                                                  const float* __restrict__ Bin,
                                                  _Float16* __restrict__ yh) {
  int g = blockIdx.x * 256 + threadIdx.x;  // over CH * B*C (chunk-major)
  int bc = g & (Bq * Cq - 1);
  int j = g >> 13;
  int c = bc & (Cq - 1);
  int b = bc >> 10;
  float dw = __expf(-__expf(td[c]));
  float eu = __expf(tf[c]);
  float Ac = Ain[(size_t)bc * CH + j];
  float Bc = Bin[(size_t)bc * CH + j];
  size_t base = (size_t)b * Tq * Cq + (size_t)j * LCH * Cq + c;
  for (int t = 0; t < LCH; ++t) {
    size_t off = base + (size_t)t * Cq;
    float ke = __expf(fminf(kraw[off], 60.0f));
    float v = vraw[off];
    float r = rraw[off];
    float kv = ke * v;
    float wkv = eu * kv + Ac;
    float wk = eu * ke + Bc + 1e-9f;
    Ac = kv + dw * Ac;
    Bc = ke + dw * Bc;
    float sig = 1.0f / (1.0f + __expf(-r));
    yh[off] = (_Float16)(sig * wkv / wk);
  }
}

// ---------------------------------------------------------------------------
extern "C" void kernel_launch(void* const* d_in, const int* in_sizes, int n_in,
                              void* d_out, int out_size, void* d_ws, size_t ws_size,
                              hipStream_t stream) {
  (void)in_sizes; (void)n_in; (void)out_size; (void)ws_size;

  const float* x  = (const float*)d_in[0];
  const float* td = (const float*)d_in[1];
  const float* tf = (const float*)d_in[2];
  const float* mk = (const float*)d_in[3];
  const float* mv = (const float*)d_in[4];
  const float* mr = (const float*)d_in[5];
  const float* Wk = (const float*)d_in[6];
  const float* Wv = (const float*)d_in[7];
  const float* Wr = (const float*)d_in[8];
  const float* Wo = (const float*)d_in[9];

  const size_t MT    = (size_t)Bq * Tq;  // 8192 rows
  const size_t nElem = MT * Cq;          // 8,388,608
  const size_t nW    = (size_t)Cq * Cq;  // 1,048,576

  // Workspace layout (~161 MB):
  _Float16* xk  = (_Float16*)d_ws;
  _Float16* xv  = xk + nElem;
  _Float16* xr  = xv + nElem;
  _Float16* WkH = xr + nElem;
  _Float16* WvH = WkH + nW;
  _Float16* WrH = WvH + nW;
  _Float16* WoH = WrH + nW;
  float* kraw = (float*)(WoH + nW);
  float* vraw = kraw + nElem;
  float* rraw = vraw + nElem;
  float* Aend = rraw + nElem;
  float* Bend = Aend + (size_t)Bq * Cq * CH;
  _Float16* yh = xk;  // xk region is dead after the k-GEMM

  cvt_w4<<<(int)((nW + 255) / 256), 256, 0, stream>>>(Wk, Wv, Wr, Wo, WkH, WvH, WrH, WoH, (int)nW);
  mix_kernel<<<(int)(nElem / 256), 256, 0, stream>>>(x, mk, mv, mr, xk, xv, xr);

  dim3 gg(Cq / 128, (unsigned)(MT / 128));
  gemm_wmma_lds<<<gg, 128, 0, stream>>>(xk, WkH, kraw, (int)MT, Cq, Cq);
  gemm_wmma_lds<<<gg, 128, 0, stream>>>(xv, WvH, vraw, (int)MT, Cq, Cq);
  gemm_wmma_lds<<<gg, 128, 0, stream>>>(xr, WrH, rraw, (int)MT, Cq, Cq);

  int nScan = Bq * Cq * CH;  // 131072
  scan_part1<<<nScan / 256, 256, 0, stream>>>(kraw, vraw, td, Aend, Bend);
  scan_part2<<<(Bq * Cq) / 256, 256, 0, stream>>>(td, Aend, Bend);
  scan_part3<<<nScan / 256, 256, 0, stream>>>(kraw, vraw, rraw, td, tf, Aend, Bend, yh);

  gemm_wmma_lds<<<gg, 128, 0, stream>>>(yh, WoH, (float*)d_out, (int)MT, Cq, Cq);
}
